// TwoLevelUNetSPConv_64785286693672
// MI455X (gfx1250) — matrix operand order repaired and checked by reading, not compiled, not run on hardware
//
#include <hip/hip_runtime.h>
#include <stdint.h>

typedef __attribute__((ext_vector_type(16))) __bf16        v16bf;
typedef __attribute__((ext_vector_type(8)))  float         v8f;
typedef __attribute__((ext_vector_type(4)))  unsigned int  u32x4;

union Frag32 { u32x4 q[2]; v16bf v; };

__device__ inline unsigned short f2bf(float f) {
  unsigned int u = __builtin_bit_cast(unsigned int, f);
  unsigned int r = (u + 0x7FFFu + ((u >> 16) & 1u)) >> 16;   // round-to-nearest-even
  return (unsigned short)r;
}
__device__ inline unsigned int packbf2(float a, float b) {
  return (unsigned int)f2bf(a) | ((unsigned int)f2bf(b) << 16);
}

// ---------------------------------------------------------------------------
// Up-conv: out[j, Cout] = sum_{c=0..2} X[3j+c, Cin] @ W[off[3j+c]]
// K-expansion: A row j has child c's features at K-slot off_c*Cin (offsets
// distinct per parent -> no collision). B = W flat [8*Cin, Cout], shared.
// ---------------------------------------------------------------------------
template<int Cin, int Cout>
__global__ __launch_bounds__(128) void upconv_kernel(
    const float* __restrict__ X, const int* __restrict__ off,
    const float* __restrict__ W, float* __restrict__ Y, int Np)
{
  constexpr int WAVES = 4;
  constexpr int KBIG  = 8 * Cin;
  constexpr int NCHUNK = KBIG / 32;          // K chunks of 32 (bf16 WMMA)
  constexpr int NCOLT  = Cout / 16;          // output column tiles
  constexpr int SLOT_CHUNKS = Cin / 32;      // K chunks per kernel-offset slot
  constexpr int WBYTES = 2 * KBIG * Cout;    // weight LDS bytes (bf16, [N][K])
  constexpr int SB     = 48 * Cin * 2;       // per-wave child staging bytes

  extern __shared__ char smem[];
  unsigned short* WT = (unsigned short*)smem;      // WT[n*KBIG + (s*Cin+i)]

  const int tid  = threadIdx.x;
  const int lane = tid & 31;
  const int wave = tid >> 5;
  const int r = lane & 15;       // matrix row (A) / column (B/D)
  const int h = lane >> 4;       // lane half

  // Stage weights to LDS, transposed to [N][K] bf16 (B frag = contiguous run).
  for (int idx = tid; idx < KBIG * Cout; idx += blockDim.x) {
    int n = idx % Cout, si = idx / Cout;           // W flat = (s*Cin+i)*Cout+n
    WT[n * KBIG + si] = f2bf(W[idx]);
  }
  __syncthreads();

  unsigned int* stg  = (unsigned int*)(smem + WBYTES + wave * (SB + 64));
  unsigned int* utbl = (unsigned int*)(smem + WBYTES + wave * (SB + 64) + SB);
  const unsigned short* stgH = (const unsigned short*)stg;

  const int nwaves = gridDim.x * WAVES;
  const int gw     = blockIdx.x * WAVES + wave;
  const int ntiles = (Np + 15) >> 4;
  const int64_t Nfine = (int64_t)Np * 3;

  for (int tile = gw; tile < ntiles; tile += nwaves) {
    const int64_t fine0 = (int64_t)tile * 48;
    const int64_t baseE = fine0 * Cin;
    const int64_t limit = (Nfine - fine0) * Cin;
    // Stage 48 contiguous child rows as bf16 (coalesced global reads).
    for (int e = lane; e < (48 * Cin) / 2; e += 32) {
      int64_t idx = (int64_t)e * 2;
      float a = (idx     < limit) ? X[baseE + idx]     : 0.f;
      float b = (idx + 1 < limit) ? X[baseE + idx + 1] : 0.f;
      stg[e] = packbf2(a, b);
    }
    // Per-row slot table: nibble s -> child index c (0..2) or 0xF.
    if (lane < 16) {
      unsigned int t = 0xFFFFFFFFu;
      int64_t f = fine0 + lane * 3;
      for (int c = 0; c < 3; ++c)
        if (f + c < Nfine) {
          int s = off[f + c] & 7;
          t = (t & ~(0xFu << (4 * s))) | ((unsigned)c << (4 * s));
        }
      utbl[lane] = t;
    }
    unsigned int trow = utbl[r];

    v8f acc[NCOLT] = {};
#pragma unroll
    for (int kc = 0; kc < NCHUNK; ++kc) {
      const int s  = kc / SLOT_CHUNKS;
      const int k0 = (kc % SLOT_CHUNKS) * 32;
      const unsigned int c = (trow >> (4 * s)) & 0xFu;
      Frag32 fa = {};
      if (c <= 2u) {  // A layout: runs at k0+h*8 and k0+16+h*8
        const unsigned short* p = stgH + ((r * 3 + (int)c) * Cin + k0 + h * 8);
        fa.q[0] = *(const u32x4*)p;
        fa.q[1] = *(const u32x4*)(p + 16);
      }
#pragma unroll
      for (int nt = 0; nt < NCOLT; ++nt) {
        // B layout: lane=col, runs at h*8 and 16+h*8
        const unsigned short* pb = WT + ((nt * 16 + r) * KBIG + kc * 32 + h * 8);
        Frag32 fb;
        fb.q[0] = *(const u32x4*)pb;
        fb.q[1] = *(const u32x4*)(pb + 16);
        acc[nt] = __builtin_amdgcn_wmma_f32_16x16x32_bf16(
            false, fa.v, false, fb.v, (short)0, acc[nt], false, false);
      }
    }
    // D layout: lane l holds rows m = v + 8*h at col l%16.
    const int j0 = tile * 16;
#pragma unroll
    for (int nt = 0; nt < NCOLT; ++nt) {
#pragma unroll
      for (int v = 0; v < 8; ++v) {
        int j = j0 + v + 8 * h;
        if (j < Np) Y[(int64_t)j * Cout + nt * 16 + r] = acc[nt][v];
      }
    }
  }
}

// ---------------------------------------------------------------------------
// Inverse conv: out[3j+c] = X[j] @ W[off[3j+c]].
// y[j] = X[j] @ Wcat[Cin, 8*Cout]; scatter only needed Cout-slices from regs.
// ---------------------------------------------------------------------------
template<int Cin, int Cout>
__global__ __launch_bounds__(128) void dnconv_kernel(
    const float* __restrict__ X, const int* __restrict__ off,
    const float* __restrict__ W, float* __restrict__ Y, int Np)
{
  constexpr int WAVES  = 4;
  constexpr int NBIG   = 8 * Cout;
  constexpr int NCOLT  = NBIG / 16;
  constexpr int NCHUNK = Cin / 32;
  constexpr int CPS    = Cout / 16;          // col tiles per slot
  constexpr int WBYTES = 2 * NBIG * Cin;
  constexpr int SB     = 16 * Cin * 2;

  extern __shared__ char smem[];
  unsigned short* WT = (unsigned short*)smem;      // WT[(s*Cout+o)*Cin + i]

  const int tid = threadIdx.x, lane = tid & 31, wave = tid >> 5;
  const int r = lane & 15, h = lane >> 4;

  for (int idx = tid; idx < 8 * Cin * Cout; idx += blockDim.x) {
    int o = idx % Cout, si = idx / Cout;           // si = s*Cin + i
    int s = si / Cin,   i = si % Cin;
    WT[(s * Cout + o) * Cin + i] = f2bf(W[idx]);
  }
  __syncthreads();

  unsigned int* stg  = (unsigned int*)(smem + WBYTES + wave * (SB + 64));
  unsigned int* utbl = (unsigned int*)(smem + WBYTES + wave * (SB + 64) + SB);
  const unsigned short* stgH = (const unsigned short*)stg;

  const int nwaves = gridDim.x * WAVES;
  const int gw     = blockIdx.x * WAVES + wave;
  const int ntiles = (Np + 15) >> 4;
  const int64_t Nfine = (int64_t)Np * 3;

  for (int tile = gw; tile < ntiles; tile += nwaves) {
    const int j0 = tile * 16;
    const int64_t baseE = (int64_t)j0 * Cin;
    const int64_t limit = ((int64_t)Np - j0) * Cin;
    for (int e = lane; e < (16 * Cin) / 2; e += 32) {
      int64_t idx = (int64_t)e * 2;
      float a = (idx     < limit) ? X[baseE + idx]     : 0.f;
      float b = (idx + 1 < limit) ? X[baseE + idx + 1] : 0.f;
      stg[e] = packbf2(a, b);
    }
    if (lane < 16) {
      unsigned int t = 0xFFFFFFFFu;
      int64_t f = (int64_t)tile * 48 + lane * 3;
      for (int c = 0; c < 3; ++c)
        if (f + c < Nfine) {
          int s = off[f + c] & 7;
          t = (t & ~(0xFu << (4 * s))) | ((unsigned)c << (4 * s));
        }
      utbl[lane] = t;
    }

    Frag32 afr[NCHUNK];
#pragma unroll
    for (int kc = 0; kc < NCHUNK; ++kc) {
      const unsigned short* p = stgH + (r * Cin + kc * 32 + h * 8);
      afr[kc].q[0] = *(const u32x4*)p;
      afr[kc].q[1] = *(const u32x4*)(p + 16);
    }
    unsigned int rowtbl[8];
#pragma unroll
    for (int v = 0; v < 8; ++v) rowtbl[v] = utbl[v + 8 * h];

#pragma unroll
    for (int nt = 0; nt < NCOLT; ++nt) {
      v8f acc = {};
#pragma unroll
      for (int kc = 0; kc < NCHUNK; ++kc) {
        const unsigned short* pb = WT + ((nt * 16 + r) * Cin + kc * 32 + h * 8);
        Frag32 fb;
        fb.q[0] = *(const u32x4*)pb;
        fb.q[1] = *(const u32x4*)(pb + 16);
        acc = __builtin_amdgcn_wmma_f32_16x16x32_bf16(
            false, afr[kc].v, false, fb.v, (short)0, acc, false, false);
      }
      const int s   = nt / CPS;
      const int col = (nt % CPS) * 16 + r;
#pragma unroll
      for (int v = 0; v < 8; ++v) {
        int m = v + 8 * h;
        unsigned int c = (rowtbl[v] >> (4 * s)) & 0xFu;
        int64_t fine = (int64_t)tile * 48 + m * 3 + (int)c;
        if (c <= 2u && fine < Nfine)
          Y[fine * Cout + col] = acc[v];
      }
    }
  }
}

// ---------------------------------------------------------------------------
extern "C" void kernel_launch(void* const* d_in, const int* in_sizes, int n_in,
                              void* d_out, int out_size, void* d_ws, size_t ws_size,
                              hipStream_t stream) {
  const float* feats = (const float*)d_in[0];
  const float* w_up1 = (const float*)d_in[1];
  const float* w_up2 = (const float*)d_in[2];
  const float* w_dn2 = (const float*)d_in[3];
  const float* w_dn1 = (const float*)d_in[4];
  const int*   off1  = (const int*)d_in[6];
  const int*   off2  = (const int*)d_in[8];

  const int N0 = in_sizes[5];       // parent1 count = fine voxels level 0
  const int N1 = in_sizes[7];       // parent2 count = level-1 voxels
  const int N2 = N1 / 3;

  float* out = (float*)d_out;       // (x_down2, x_down1, x_up2, x_up1) flat
  float* xd2 = out;
  float* xd1 = out + (size_t)N0 * 32;
  float* xu2 = xd1 + (size_t)N1 * 64;
  float* xu1 = xu2 + (size_t)N2 * 128;

  auto blocksFor = [](int Np) {
    int tiles = (Np + 15) / 16;
    int b = (tiles + 3) / 4;        // 4 waves per block
    if (b > 1536) b = 1536;
    if (b < 1) b = 1;
    return b;
  };

  {
    size_t sm = 2 * 8 * 32 * 64 + 4 * (48 * 32 * 2 + 64);            // ~45 KB
    upconv_kernel<32, 64><<<blocksFor(N1), 128, sm, stream>>>(feats, off1, w_up1, xu1, N1);
  }
  {
    size_t sm = 2 * 8 * 64 * 128 + 4 * (48 * 64 * 2 + 64);           // ~153 KB
    upconv_kernel<64, 128><<<blocksFor(N2), 128, sm, stream>>>(xu1, off2, w_up2, xu2, N2);
  }
  {
    size_t sm = 2 * 8 * 128 * 64 + 4 * (16 * 128 * 2 + 64);          // ~145 KB
    dnconv_kernel<128, 64><<<blocksFor(N2), 128, sm, stream>>>(xu2, off2, w_dn2, xd1, N2);
  }
  {
    size_t sm = 2 * 8 * 64 * 32 + 4 * (16 * 64 * 2 + 64);            // ~41 KB
    dnconv_kernel<64, 32><<<blocksFor(N1), 128, sm, stream>>>(xd1, off1, w_dn1, xd2, N1);
  }
}
